// WindowAttention_2284922601547
// MI455X (gfx1250) — matrix operand order repaired and checked by reading, not compiled, not run on hardware
//
#include <hip/hip_runtime.h>

typedef __attribute__((ext_vector_type(16))) _Float16 v16h;
typedef __attribute__((ext_vector_type(8)))  _Float16 v8h;
typedef __attribute__((ext_vector_type(8)))  float    v8f;

#define NH      12
#define HD      32
#define CC      384
#define NQKV    1152
#define WIN     7
#define SHIFTS  3
#define HH      56
#define LW      49
#define KTILES  12      // 384 / 32
#define QKV_NT  72      // 1152 / 16
#define PROJ_NT 24      // 384 / 16

// Build a v16h A-operand for v_wmma_f32_16x16x32_f16 from LDS/row-major memory.
// Contiguous K layout: lanes 0-15 hold K 0..15, lanes 16-31 hold K 16..31.
// Caller pre-applies row*stride + kbase + ((lane>>4)<<3); we add another 8*(lane>>4) and fetch [0..15].
__device__ __forceinline__ v16h load_a16(const _Float16* p) {
  p += ((threadIdx.x >> 4) & 1) << 3;
  v8h lo = *(const v8h*)p;
  v8h hi = *(const v8h*)(p + 8);
  return __builtin_shufflevector(lo, hi, 0,1,2,3,4,5,6,7,8,9,10,11,12,13,14,15);
}

__device__ __forceinline__ v8f wmma16(v16h a, v16h b, v8f c) {
  return __builtin_amdgcn_wmma_f32_16x16x32_f16(false, a, false, b, (short)0, c, false, false);
}

// Pre-swizzle a KxN f32 weight matrix into f16 WMMA-B tile layout:
// tile t = kt*(N/16)+nt holds 512 halfs; element [lane*16+i] =
//   W[kt*32 + (lane>>4)*16 + i][nt*16 + (lane&15)]  -> contiguous v16h per lane.
__global__ void swizzle_b(const float* __restrict__ W, _Float16* __restrict__ out, int N) {
  int idx = blockIdx.x * blockDim.x + threadIdx.x;
  if (idx >= CC * N) return;
  int i     = idx & 15;
  int lane  = (idx >> 4) & 31;
  int t     = idx >> 9;
  int ntile = N >> 4;
  int kt = t / ntile;
  int nt = t - kt * ntile;
  int row = kt * 32 + ((lane >> 4) << 4) + i;
  int col = nt * 16 + (lane & 15);
  out[idx] = (_Float16)W[(size_t)row * N + col];
}

__global__ __launch_bounds__(384, 1)
void swin_window_attn(const float* __restrict__ x,
                      const float* __restrict__ b_qkv,
                      const float* __restrict__ b_proj,
                      const float* __restrict__ bias_tab,   // (169, 12)
                      const _Float16* __restrict__ wqkv,    // swizzled [12*72][512]
                      const _Float16* __restrict__ wproj,   // swizzled [12*24][512]
                      float* __restrict__ out)
{
  // 149,248 B of LDS total (safe even if CU-mode halves the 320KB WGP pool)
  __shared__ __align__(32) _Float16 s_qk[LW * 768];     // Q (cols 0-383, pre-scaled) | K (384-767); Q cols reused for attn-out
  __shared__ __align__(32) _Float16 s_xv[CC * 64];      // phase1: X window [l*384+c]; phase2: V^T [d*64+slot]
  __shared__ __align__(32) _Float16 s_p [NH * 16 * 64]; // per-head softmax tile (16 queries x 64 key-slots)
  __shared__ int s_ids[64];

  const int tid  = threadIdx.x;
  const int wv   = tid >> 5;       // wave id 0..11
  const int lane = tid & 31;
  const int lh   = lane >> 4;      // half-wave select
  const int lm   = lane & 15;

  const int bn = blockIdx.x;       // window id 0..4095
  const int b  = bn >> 6;
  const int wi = bn & 63;
  const int wh = wi >> 3;
  const int ww = wi & 7;

  const int MB[4] = {0, 16, 32, 33};  // M-tile row bases (last tile overlaps rows 33-48)
  const int SB[4] = {0, 16, 32, 48};  // key-slot bases for V^T / P columns

  // ---- Phase 0: gather shifted window -> LDS (f32 -> f16), region ids for mask ----
  #pragma unroll 2
  for (int l = 0; l < LW; ++l) {
    int lr = l / WIN, lc = l % WIN;
    int h = wh * WIN + lr + SHIFTS; if (h >= HH) h -= HH;   // roll(x, -s)
    int w = ww * WIN + lc + SHIFTS; if (w >= HH) w -= HH;
    float v = x[(((size_t)b * HH + h) * HH + w) * CC + tid];
    s_xv[l * CC + tid] = (_Float16)v;
  }
  if (tid < LW) {
    int lr = tid / WIN, lc = tid % WIN;
    int gh = wh * WIN + lr, gw = ww * WIN + lc;
    int rh = (gh < 49) ? 0 : ((gh < 52) ? 1 : 2);
    int rw = (gw < 49) ? 0 : ((gw < 52) ? 1 : 2);
    s_ids[tid] = rh * 3 + rw;
  }
  __syncthreads();

  const float scale = 0.17677669529663687f;  // 32^-0.5, folded into Q

  // ---- Phase 1a: Q,K GEMM. Wave wv owns output cols [wv*64, wv*64+64) of 0..767 ----
  #pragma unroll 1
  for (int nt = 0; nt < 4; ++nt) {
    const int gnt = wv * 4 + nt;                 // global n-tile 0..47
    const int col = gnt * 16 + lm;
    const float bq = b_qkv[col];
    const bool  isQ = (gnt < 24);
    v8f acc[4] = {};
    #pragma unroll 1
    for (int kt = 0; kt < KTILES; ++kt) {
      v16h bmat = *(const v16h*)(wqkv + ((size_t)(kt * QKV_NT + gnt) << 9) + (lane << 4));
      #pragma unroll
      for (int mt = 0; mt < 4; ++mt) {
        v16h amat = load_a16(s_xv + (MB[mt] + lm) * CC + kt * 32 + (lh << 3));
        acc[mt] = wmma16(amat, bmat, acc[mt]);
      }
    }
    #pragma unroll
    for (int mt = 0; mt < 4; ++mt)
      for (int r = 0; r < 8; ++r) {
        float v = acc[mt][r] + bq;
        if (isQ) v *= scale;
        s_qk[(MB[mt] + (lh << 3) + r) * 768 + col] = (_Float16)v;
      }
  }

  // ---- Phase 1b: V GEMM into registers (cols 768 + wv*32 .. +32) ----
  v8f vacc[2][4] = {};
  #pragma unroll 1
  for (int kt = 0; kt < KTILES; ++kt) {
    const int g0 = 48 + wv * 2;
    v16h bmat0 = *(const v16h*)(wqkv + ((size_t)(kt * QKV_NT + g0    ) << 9) + (lane << 4));
    v16h bmat1 = *(const v16h*)(wqkv + ((size_t)(kt * QKV_NT + g0 + 1) << 9) + (lane << 4));
    #pragma unroll
    for (int mt = 0; mt < 4; ++mt) {
      v16h amat = load_a16(s_xv + (MB[mt] + lm) * CC + kt * 32 + (lh << 3));
      vacc[0][mt] = wmma16(amat, bmat0, vacc[0][mt]);
      vacc[1][mt] = wmma16(amat, bmat1, vacc[1][mt]);
    }
  }
  _Float16 vh[2][4][8];
  #pragma unroll
  for (int n2 = 0; n2 < 2; ++n2) {
    float bv = b_qkv[768 + wv * 32 + n2 * 16 + lm];
    for (int mt = 0; mt < 4; ++mt)
      for (int r = 0; r < 8; ++r)
        vh[n2][mt][r] = (_Float16)(vacc[n2][mt][r] + bv);
  }
  __syncthreads();  // all X reads done; s_xv becomes V^T
  #pragma unroll
  for (int n2 = 0; n2 < 2; ++n2) {
    int d = wv * 32 + n2 * 16 + lm;
    for (int mt = 0; mt < 4; ++mt)
      for (int r = 0; r < 8; ++r)
        s_xv[d * 64 + SB[mt] + (lh << 3) + r] = vh[n2][mt][r];
  }
  __syncthreads();

  // ---- Phase 2: attention, wave = head ----
  const int hh = wv;
  const _Float16* qbase = s_qk + hh * HD;
  const _Float16* kbase = s_qk + 384 + hh * HD;
  _Float16* ph = s_p + hh * 16 * 64;
  _Float16 oh23[2][2][8];   // deferred outputs for the overlapping row-tiles 2 and 3

  #pragma unroll
  for (int mt = 0; mt < 4; ++mt) {
    v16h qa = load_a16(qbase + (MB[mt] + lm) * 768 + (lh << 3));
    v8f sv[4];
    #pragma unroll
    for (int nt = 0; nt < 4; ++nt) {
      int key = (nt < 3 ? nt * 16 : 33) + lm;   // tile 3 covers keys 33..48
      v16h kb = *(const v16h*)(kbase + key * 768 + (lh << 4));  // K^T B-operand: contiguous
      v8f z = {};
      sv[nt] = wmma16(qa, kb, z);
    }
    // bias + shift-mask; kill duplicate key columns in the overlapped tile
    #pragma unroll
    for (int nt = 0; nt < 4; ++nt) {
      bool valid = (nt < 3) || (lm == 15);
      int key = (nt < 3) ? (nt * 16 + lm) : 48;
      int i2 = key / WIN, j2 = key % WIN;
      int idn = s_ids[key];
      for (int r = 0; r < 8; ++r) {
        float v = -1e30f;
        if (valid) {
          int m = MB[mt] + (lh << 3) + r;       // query index, always <= 48
          int i1 = m / WIN, j1 = m % WIN;
          int rel = (i1 - i2 + 6) * 13 + (j1 - j2 + 6);
          v = sv[nt][r] + bias_tab[rel * NH + hh];
          if (s_ids[m] != idn) v -= 100.f;
        }
        sv[nt][r] = v;
      }
    }
    // softmax over the 64 key-slots (masked ones -> 0)
    #pragma unroll
    for (int r = 0; r < 8; ++r) {
      float mx = fmaxf(fmaxf(sv[0][r], sv[1][r]), fmaxf(sv[2][r], sv[3][r]));
      for (int off = 1; off < 16; off <<= 1) mx = fmaxf(mx, __shfl_xor(mx, off, 32));
      float p0 = __expf(sv[0][r] - mx);
      float p1 = __expf(sv[1][r] - mx);
      float p2 = __expf(sv[2][r] - mx);
      float p3 = __expf(sv[3][r] - mx);
      float sum = p0 + p1 + p2 + p3;
      for (int off = 1; off < 16; off <<= 1) sum += __shfl_xor(sum, off, 32);
      float inv = 1.f / sum;
      int rr = (lh << 3) + r;
      ph[rr * 64 + lm +  0] = (_Float16)(p0 * inv);
      ph[rr * 64 + lm + 16] = (_Float16)(p1 * inv);
      ph[rr * 64 + lm + 32] = (_Float16)(p2 * inv);
      ph[rr * 64 + lm + 48] = (_Float16)(p3 * inv);
    }
    // P @ V  (K dim = 64 key-slots, V^T rows contiguous)
    v8f ov[2] = {};
    #pragma unroll
    for (int k2 = 0; k2 < 2; ++k2) {
      v16h pa = load_a16(ph + lm * 64 + k2 * 32 + (lh << 3));
      for (int n2 = 0; n2 < 2; ++n2) {
        v16h vb = *(const v16h*)(s_xv + (hh * 32 + n2 * 16 + lm) * 64 + k2 * 32 + (lh << 4));
        ov[n2] = wmma16(pa, vb, ov[n2]);
      }
    }
    if (mt < 2) {
      // rows 0-31 never alias the Q rows still to be read by tiles 2/3 -> store now
      #pragma unroll
      for (int n2 = 0; n2 < 2; ++n2)
        for (int r = 0; r < 8; ++r)
          s_qk[(MB[mt] + (lh << 3) + r) * 768 + hh * 32 + n2 * 16 + lm] = (_Float16)ov[n2][r];
    } else {
      #pragma unroll
      for (int n2 = 0; n2 < 2; ++n2)
        for (int r = 0; r < 8; ++r)
          oh23[mt - 2][n2][r] = (_Float16)ov[n2][r];
    }
  }
  // flush the deferred overlapping tiles (Q rows 32-48 fully consumed now)
  #pragma unroll
  for (int mt = 2; mt < 4; ++mt)
    for (int n2 = 0; n2 < 2; ++n2)
      for (int r = 0; r < 8; ++r)
        s_qk[(MB[mt] + (lh << 3) + r) * 768 + hh * 32 + n2 * 16 + lm] = oh23[mt - 2][n2][r];
  __syncthreads();

  // ---- Phase 3: proj GEMM + reverse-shift scatter to HBM ----
  #pragma unroll 1
  for (int nt = 0; nt < 2; ++nt) {
    int gnt = wv * 2 + nt;
    int col = gnt * 16 + lm;
    v8f acc[4] = {};
    #pragma unroll 1
    for (int kt = 0; kt < KTILES; ++kt) {
      v16h bmat = *(const v16h*)(wproj + ((size_t)(kt * PROJ_NT + gnt) << 9) + (lane << 4));
      #pragma unroll
      for (int mt = 0; mt < 4; ++mt) {
        v16h amat = load_a16(s_qk + (MB[mt] + lm) * 768 + kt * 32 + (lh << 3));
        acc[mt] = wmma16(amat, bmat, acc[mt]);
      }
    }
    float bp = b_proj[col];
    #pragma unroll
    for (int mt = 0; mt < 4; ++mt)
      for (int r = 0; r < 8; ++r) {
        int m = MB[mt] + (lh << 3) + r;         // <= 48 (duplicates write same value)
        int lr = m / WIN, lc = m % WIN;
        int ho = wh * WIN + lr + SHIFTS; if (ho >= HH) ho -= HH;   // roll(out, +s)
        int wo = ww * WIN + lc + SHIFTS; if (wo >= HH) wo -= HH;
        out[(((size_t)b * HH + ho) * HH + wo) * CC + col] = acc[mt][r] + bp;
      }
  }
}

extern "C" void kernel_launch(void* const* d_in, const int* in_sizes, int n_in,
                              void* d_out, int out_size, void* d_ws, size_t ws_size,
                              hipStream_t stream) {
  (void)in_sizes; (void)n_in; (void)out_size; (void)ws_size;
  const float* x      = (const float*)d_in[0];
  const float* w_qkv  = (const float*)d_in[1];
  const float* b_qkv  = (const float*)d_in[2];
  const float* w_proj = (const float*)d_in[3];
  const float* b_proj = (const float*)d_in[4];
  const float* btab   = (const float*)d_in[5];

  _Float16* wqkv_sw  = (_Float16*)d_ws;                      // 884,736 B
  _Float16* wproj_sw = wqkv_sw + (size_t)CC * NQKV;          // 294,912 B

  swizzle_b<<<(CC * NQKV + 255) / 256, 256, 0, stream>>>(w_qkv, wqkv_sw, NQKV);
  swizzle_b<<<(CC * CC   + 255) / 256, 256, 0, stream>>>(w_proj, wproj_sw, CC);
  swin_window_attn<<<4096, 384, 0, stream>>>(x, b_qkv, b_proj, btab,
                                             wqkv_sw, wproj_sw, (float*)d_out);
}
